// Conv6_BNN_81973745811628
// MI455X (gfx1250) — matrix operand order repaired and checked by reading, not compiled, not run on hardware
//
#include <hip/hip_runtime.h>
#include <hip/hip_bf16.h>

// ---------------------------------------------------------------------------
// Conv6 BNN forward on gfx1250 (MI455X).
// Compute-bound (~315 GFLOP vs 10s-of-us of HBM traffic at 23.3 TB/s): all
// convs/FCs run as implicit GEMM on v_wmma_f32_16x16x32_bf16 (wave32 WMMA).
// Weights (and FC activations) stream into LDS with
// global_load_async_to_lds_b128 (ASYNCcnt + s_wait_asynccnt), the im2col
// activation gather uses shift/mask addressing (all dims power-of-two;
// layer-0 Cin padded 3->4). Block tile 128x128x32, 8 waves, each wave
// 32x64 = 2x4 WMMA tiles.
// ---------------------------------------------------------------------------

typedef __bf16 bf16_t;
typedef __attribute__((ext_vector_type(16))) __bf16 v16bf;
typedef __attribute__((ext_vector_type(8)))  float  v8f;

#define BM 128
#define BN 128
#define BK 32
#define A_STRIDE (BK + 8)   // 40 bf16 = 80 B rows (16B-multiple, bank-skewed)
#define B_STRIDE (BN + 8)   // 136 bf16 = 272 B rows (16B-multiple)

// async copy: 16 bytes global -> LDS, tracked on ASYNCcnt
__device__ __forceinline__ void async_ld_b128(unsigned ldsAddr, const void* g) {
    asm volatile("global_load_async_to_lds_b128 %0, %1, off"
                 :: "v"(ldsAddr), "v"(g) : "memory");
}
__device__ __forceinline__ void wait_async0() {
    asm volatile("s_wait_asynccnt 0x0" ::: "memory");
}
// LDS byte offset of a __shared__ pointer (flat->LDS truncation, ISA 10.2)
__device__ __forceinline__ unsigned lds_off(const void* p) {
    return (unsigned)(uintptr_t)p;
}

// ---------------------------------------------------------------------------
// Implicit-GEMM conv (3x3 pad1 stride1, NHWC) / 1x1 FC, bf16 WMMA, f32 out.
// M = Nb*H*W output pixels, K = (9|1)<<cinLog2 with k=(ky*3+kx)*Cin+ci.
// Weight buffer is padded to [32-mult K][CoutPad] and zero-filled, so the
// B-tile async loads are unconditional.
// ---------------------------------------------------------------------------
__global__ __launch_bounds__(256) void conv_wmma_kernel(
    const bf16_t* __restrict__ act, const bf16_t* __restrict__ wgt,
    float* __restrict__ out,
    int M, int Kdim, int Cout, int CoutPad,
    int cinLog2, int wLog2, int hwLog2, int is3x3)
{
    __shared__ __align__(16) bf16_t Asm[BM][A_STRIDE];  // [row in M-tile][k]
    __shared__ __align__(16) bf16_t Bsm[BK][B_STRIDE];  // [k][col in N-tile]

    const int m0   = blockIdx.x * BM;
    const int n0   = blockIdx.y * BN;
    const int tid  = threadIdx.x;
    const int lane = tid & 31;
    const int wave = tid >> 5;   // 0..7
    const int wm   = wave >> 1;  // 0..3 -> 32-row strip
    const int wn   = wave & 1;   // 0..1 -> 64-col strip
    const int l    = lane & 15;
    const int hi   = lane >> 4;

    const int Wd     = 1 << wLog2;
    const int Hd     = 1 << (hwLog2 - wLog2);
    const int ciMask = (1 << cinLog2) - 1;
    const int hwMask = (1 << hwLog2) - 1;

    v8f acc[2][4];
#pragma unroll
    for (int i = 0; i < 2; ++i)
#pragma unroll
        for (int j = 0; j < 4; ++j)
#pragma unroll
            for (int e = 0; e < 8; ++e) acc[i][j][e] = 0.0f;

    const int ksteps = (Kdim + BK - 1) / BK;
    for (int kt = 0; kt < ksteps; ++kt) {
        const int kb = kt * BK;

        // ---- B tile (BK x BN): async b128 copies, 2 chunks/thread ----
        // 512 chunks of 8 bf16; buffer is padded/zero-filled so no guards.
#pragma unroll
        for (int i = 0; i < 2; ++i) {
            int seg  = tid + i * 256;       // 0..511
            int kk   = seg >> 4;            // 0..31
            int nseg = (seg & 15) << 3;     // 0,8,...,120
            const bf16_t* g = wgt + (size_t)(kb + kk) * CoutPad + n0 + nseg;
            async_ld_b128(lds_off(&Bsm[kk][nseg]), g);
        }

        if (is3x3) {
            // ---- A tile via im2col gather (shift/mask): 16 elems/thread ----
#pragma unroll
            for (int i = 0; i < 16; ++i) {
                int e  = tid + i * 256;
                int mi = e >> 5;
                int kk = e & 31;
                int m  = m0 + mi;
                int k  = kb + kk;
                bf16_t v = (bf16_t)0.0f;
                if (m < M && k < Kdim) {
                    int r  = k >> cinLog2;       // filter tap 0..8
                    int ci = k & ciMask;
                    int ky = (r * 43) >> 7;      // r/3 for r in 0..8
                    int kx = r - 3 * ky;
                    int n   = m >> hwLog2;
                    int rem = m & hwMask;
                    int y   = rem >> wLog2;
                    int x   = rem & (Wd - 1);
                    int iy = y + ky - 1, ix = x + kx - 1;
                    if (iy >= 0 && iy < Hd && ix >= 0 && ix < Wd) {
                        size_t pix = ((size_t)((n << (hwLog2 - wLog2)) + iy)
                                      << wLog2) + ix;
                        v = act[(pix << cinLog2) + ci];
                    }
                }
                Asm[mi][kk] = v;
            }
        } else {
            // ---- 1x1 A tile is a dense 2D slab: async b128, 2/thread ----
#pragma unroll
            for (int i = 0; i < 2; ++i) {
                int seg  = tid + i * 256;       // 0..511
                int mi   = seg >> 2;            // 0..127
                int kseg = (seg & 3) << 3;      // 0,8,16,24
                if (m0 + mi < M && kb + kseg < Kdim) {
                    const bf16_t* g =
                        act + (((size_t)(m0 + mi)) << cinLog2) + kb + kseg;
                    async_ld_b128(lds_off(&Asm[mi][kseg]), g);
                }
            }
        }

        wait_async0();          // our async writes are in LDS
        __syncthreads();        // everyone's are visible

        // ---- build fragments per ISA 7.12.2 layouts, issue 8 WMMAs ----
        v16bf afrag[2], bfrag[4];
#pragma unroll
        for (int tm = 0; tm < 2; ++tm) {
            int row = wm * 32 + tm * 16 + l;     // both lane halves: M = lane%16
#pragma unroll
            for (int vv = 0; vv < 8; ++vv) {
                int kp = (vv < 4) ? (hi * 8 + 2 * vv)
                                  : (16 + hi * 8 + 2 * (vv - 4));
                afrag[tm][2 * vv]     = Asm[row][kp];
                afrag[tm][2 * vv + 1] = Asm[row][kp + 1];
            }
        }
#pragma unroll
        for (int tn = 0; tn < 4; ++tn) {
            int col = wn * 64 + tn * 16 + l;     // N = lane%16
#pragma unroll
            for (int vv = 0; vv < 8; ++vv) {
                int kr = (vv < 4) ? (hi * 8 + 2 * vv)
                                  : (16 + hi * 8 + 2 * (vv - 4));  // same K order as A
                bfrag[tn][2 * vv]     = Bsm[kr][col];
                bfrag[tn][2 * vv + 1] = Bsm[kr + 1][col];
            }
        }
#pragma unroll
        for (int tm = 0; tm < 2; ++tm)
#pragma unroll
            for (int tn = 0; tn < 4; ++tn)
                acc[tm][tn] = __builtin_amdgcn_wmma_f32_16x16x32_bf16(
                    false, afrag[tm], false, bfrag[tn],
                    (short)0, acc[tm][tn], false, false);

        __syncthreads();
    }

    // ---- store C: VGPR r -> row r + 8*hi, col = lane%16 ----
#pragma unroll
    for (int tm = 0; tm < 2; ++tm)
#pragma unroll
        for (int tn = 0; tn < 4; ++tn) {
            int mb = m0 + wm * 32 + tm * 16 + hi * 8;
            int nb = n0 + wn * 64 + tn * 16 + l;
            if (nb < Cout) {
#pragma unroll
                for (int r = 0; r < 8; ++r) {
                    int m = mb + r;
                    if (m < M) out[(size_t)m * Cout + nb] = acc[tm][tn][r];
                }
            }
        }
}

// ---- per-channel training-mode BN stats: mean & rsqrt(var+eps) ----
__global__ __launch_bounds__(256) void stats_kernel(
    const float* __restrict__ x, float* __restrict__ mean,
    float* __restrict__ rstd, int M, int C)
{
    const int c = blockIdx.x;
    float s = 0.0f, s2 = 0.0f;
    for (int i = threadIdx.x; i < M; i += 256) {
        float v = x[(size_t)i * C + c];
        s += v; s2 += v * v;
    }
    __shared__ float sh[256], sh2[256];
    sh[threadIdx.x] = s; sh2[threadIdx.x] = s2;
    __syncthreads();
    for (int st = 128; st > 0; st >>= 1) {
        if (threadIdx.x < st) {
            sh[threadIdx.x]  += sh[threadIdx.x + st];
            sh2[threadIdx.x] += sh2[threadIdx.x + st];
        }
        __syncthreads();
    }
    if (threadIdx.x == 0) {
        float mu  = sh[0] / (float)M;
        float var = sh2[0] / (float)M - mu * mu;
        mean[c] = mu;
        rstd[c] = rsqrtf(var + 1e-5f);
    }
}

// ---- fused BN + sign -> bf16 (+1/-1/0) NHWC activation ----
__global__ __launch_bounds__(256) void bnsign_kernel(
    const float* __restrict__ x, const float* __restrict__ mean,
    const float* __restrict__ rstd, const float* __restrict__ g,
    const float* __restrict__ b, bf16_t* __restrict__ out,
    size_t total, int C)
{
    size_t i = (size_t)blockIdx.x * 256 + threadIdx.x;
    if (i >= total) return;
    int c = (int)(i % (size_t)C);
    float y = (x[i] - mean[c]) * rstd[c] * g[c] + b[c];
    float s = (y > 0.0f) ? 1.0f : ((y < 0.0f) ? -1.0f : 0.0f);
    out[i] = (bf16_t)s;
}

// ---- 2x2 maxpool on fp32 NHWC ----
__global__ __launch_bounds__(256) void pool_kernel(
    const float* __restrict__ in, float* __restrict__ out,
    int Nb, int H, int W, int C)
{
    const int Ho = H >> 1, Wo = W >> 1;
    size_t total = (size_t)Nb * Ho * Wo * C;
    size_t i = (size_t)blockIdx.x * 256 + threadIdx.x;
    if (i >= total) return;
    int c = (int)(i % C);
    size_t t = i / C;
    int x = (int)(t % Wo); t /= Wo;
    int y = (int)(t % Ho);
    int n = (int)(t / Ho);
    size_t base = ((size_t)(n * H + 2 * y) * W + 2 * x) * C + c;
    size_t rs = (size_t)W * C;
    float a0 = in[base],      a1 = in[base + C];
    float a2 = in[base + rs], a3 = in[base + rs + C];
    out[i] = fmaxf(fmaxf(a0, a1), fmaxf(a2, a3));
}

// ---- conv weight prep: OIHW fp32 -> padded [KdimPad][CoutPad] bf16,
// k=(ky*3+kx)*CinPad+ci; zero-fills Cin pad, K pad, Cout pad. ----
__global__ __launch_bounds__(256) void prep_conv_w(
    const float* __restrict__ src, bf16_t* __restrict__ dst,
    int Cin, int CinPad, int Cout, int CoutPad, int KdimPad)
{
    size_t total = (size_t)KdimPad * CoutPad;
    size_t i = (size_t)blockIdx.x * 256 + threadIdx.x;
    if (i >= total) return;
    int k  = (int)(i / CoutPad);
    int co = (int)(i - (size_t)k * CoutPad);
    bf16_t v = (bf16_t)0.0f;
    if (k < 9 * CinPad && co < Cout) {
        int r  = k / CinPad;
        int ci = k - r * CinPad;
        int ky = r / 3, kx = r - ky * 3;
        if (ci < Cin)
            v = (bf16_t)src[((size_t)(co * Cin + ci) * 3 + ky) * 3 + kx];
    }
    dst[i] = v;
}

// ---- FC weight prep: [Cout][Cin_ref] fp32 -> padded [Cin][CoutPad] bf16.
// For FC1 the reference flattens NCHW (c*16 + y*4 + x); our acts are NHWC
// ((y*4+x)*512 + c), so fold the permutation into the weight layout. ----
__global__ __launch_bounds__(256) void prep_fc_w(
    const float* __restrict__ src, bf16_t* __restrict__ dst,
    int Cin, int Cout, int CoutPad, int permute)
{
    size_t total = (size_t)Cin * CoutPad;
    size_t i = (size_t)blockIdx.x * 256 + threadIdx.x;
    if (i >= total) return;
    int k  = (int)(i / CoutPad);
    int co = (int)(i - (size_t)k * CoutPad);
    bf16_t v = (bf16_t)0.0f;
    if (co < Cout) {
        int cin_ref = k;
        if (permute) {
            int c = k & 511;       // NHWC channel
            int p = k >> 9;        // y*4 + x
            cin_ref = c * 16 + p;  // NCHW flat index
        }
        v = (bf16_t)src[(size_t)co * Cin + cin_ref];
    }
    dst[i] = v;
}

// ---- input fp32 NCHW (C=3) -> bf16 NHWC padded to C=4 (ch 3 = 0) ----
__global__ __launch_bounds__(256) void convert_input_kernel(
    const float* __restrict__ x, bf16_t* __restrict__ act,
    int Nb, int H, int W)
{
    size_t total = (size_t)Nb * H * W * 4;
    size_t i = (size_t)blockIdx.x * 256 + threadIdx.x;
    if (i >= total) return;
    int c = (int)(i & 3);
    size_t t = i >> 2;
    int xw = (int)(t % W); t /= W;
    int y  = (int)(t % H);
    int n  = (int)(t / H);
    float v = (c < 3) ? x[((size_t)(n * 3 + c) * H + y) * W + xw] : 0.0f;
    act[i] = (bf16_t)v;
}

// ---------------------------------------------------------------------------
extern "C" void kernel_launch(void* const* d_in, const int* in_sizes, int n_in,
                              void* d_out, int out_size, void* d_ws, size_t ws_size,
                              hipStream_t stream) {
    (void)in_sizes; (void)n_in; (void)out_size; (void)ws_size;
    const float* x = (const float*)d_in[0];
    const float* conv_w[6]; for (int i = 0; i < 6; ++i) conv_w[i] = (const float*)d_in[1 + i];
    const float* fc_w[3];   for (int i = 0; i < 3; ++i) fc_w[i]   = (const float*)d_in[7 + i];
    const float* bn_g[8];   for (int i = 0; i < 8; ++i) bn_g[i]   = (const float*)d_in[10 + i];
    const float* bn_b[8];   for (int i = 0; i < 8; ++i) bn_b[i]   = (const float*)d_in[18 + i];

    // workspace layout (~319 MB)
    char* ws = (char*)d_ws;
    const size_t ACT_BYTES  = 256ull * 32 * 32 * 128 * 2;  //  67,108,864
    const size_t CONV_BYTES = 256ull * 32 * 32 * 128 * 4;  // 134,217,728
    const size_t POOL_BYTES = 256ull * 16 * 16 * 128 * 4;  //  33,554,432
    const size_t WGT_BYTES  = 8192ull * 1024 * 2;          //  16,777,216
    bf16_t* actA    = (bf16_t*)ws;
    bf16_t* actB    = (bf16_t*)(ws + ACT_BYTES);
    float*  convOut = (float*)(ws + 2 * ACT_BYTES);
    float*  poolOut = (float*)(ws + 2 * ACT_BYTES + CONV_BYTES);
    bf16_t* wgt     = (bf16_t*)(ws + 2 * ACT_BYTES + CONV_BYTES + POOL_BYTES);
    float*  meanBuf = (float*)(ws + 2 * ACT_BYTES + CONV_BYTES + POOL_BYTES + WGT_BYTES);
    float*  rstdBuf = meanBuf + 1024;

    const int Nb = 256;
    auto blocks1d = [](size_t total) { return (unsigned)((total + 255) / 256); };

    // input fp32 NCHW -> bf16 NHWC (padded to 4 channels)
    convert_input_kernel<<<blocks1d(256ull * 32 * 32 * 4), 256, 0, stream>>>(
        x, actA, Nb, 32, 32);

    // Cin below is the padded (power-of-two) input-channel count.
    struct Layer { int Cin, Cout, H, W, cinL, wL, hwL, pool, is3, widx, bni; };
    const Layer L[8] = {
        {   4,  128, 32, 32,  2, 5, 10, 0, 1, 0, 0 },
        { 128,  128, 32, 32,  7, 5, 10, 1, 1, 1, 1 },
        { 128,  256, 16, 16,  7, 4,  8, 0, 1, 2, 2 },
        { 256,  256, 16, 16,  8, 4,  8, 1, 1, 3, 3 },
        { 256,  512,  8,  8,  8, 3,  6, 0, 1, 4, 4 },
        { 512,  512,  8,  8,  9, 3,  6, 1, 1, 5, 5 },
        {8192, 1024,  1,  1, 13, 0,  0, 0, 0, 0, 6 },   // fc1 (permuted prep)
        {1024, 1024,  1,  1, 10, 0,  0, 0, 0, 1, 7 },   // fc2
    };

    bf16_t* cur = actA;
    bf16_t* nxt = actB;
    for (int li = 0; li < 8; ++li) {
        const Layer& c = L[li];
        int Kdim    = (c.is3 ? 9 : 1) << c.cinL;
        int KdimPad = (Kdim + BK - 1) / BK * BK;
        int CoutPad = (c.Cout + BN - 1) / BN * BN;
        // weight prep (zero-padded [KdimPad][CoutPad] bf16)
        if (c.is3) {
            int cinReal = (li == 0) ? 3 : c.Cin;
            size_t wtot = (size_t)KdimPad * CoutPad;
            prep_conv_w<<<blocks1d(wtot), 256, 0, stream>>>(
                conv_w[c.widx], wgt, cinReal, c.Cin, c.Cout, CoutPad, KdimPad);
        } else {
            size_t wtot = (size_t)Kdim * CoutPad;
            prep_fc_w<<<blocks1d(wtot), 256, 0, stream>>>(
                fc_w[c.widx], wgt, c.Cin, c.Cout, CoutPad, li == 6 ? 1 : 0);
        }
        // conv / FC as WMMA implicit GEMM
        int M = Nb * c.H * c.W;
        dim3 grid((M + BM - 1) / BM, (c.Cout + BN - 1) / BN);
        conv_wmma_kernel<<<grid, 256, 0, stream>>>(cur, wgt, convOut,
                                                   M, Kdim, c.Cout, CoutPad,
                                                   c.cinL, c.wL, c.hwL, c.is3);
        // optional maxpool (reference order: conv -> pool -> BN -> sign)
        const float* bnSrc = convOut;
        int Ms = M;
        if (c.pool) {
            size_t ptot = (size_t)Nb * (c.H / 2) * (c.W / 2) * c.Cout;
            pool_kernel<<<blocks1d(ptot), 256, 0, stream>>>(convOut, poolOut,
                                                            Nb, c.H, c.W, c.Cout);
            bnSrc = poolOut;
            Ms = M / 4;
        }
        // BN stats + fused BN+sign -> next bf16 activations
        stats_kernel<<<c.Cout, 256, 0, stream>>>(bnSrc, meanBuf, rstdBuf, Ms, c.Cout);
        size_t atot = (size_t)Ms * c.Cout;
        bnsign_kernel<<<blocks1d(atot), 256, 0, stream>>>(bnSrc, meanBuf, rstdBuf,
                                                          bn_g[c.bni], bn_b[c.bni],
                                                          nxt, atot, c.Cout);
        bf16_t* t = cur; cur = nxt; nxt = t;
    }

    // final FC: 1024 -> 10 (CoutPad 128), no BN/sign; output [256][10] == d_out
    prep_fc_w<<<blocks1d(1024ull * 128), 256, 0, stream>>>(fc_w[2], wgt,
                                                           1024, 10, 128, 0);
    dim3 gridF((256 + BM - 1) / BM, 1);
    conv_wmma_kernel<<<gridF, 256, 0, stream>>>(cur, wgt, (float*)d_out,
                                                256, 1024, 10, 128, 10, 0, 0, 0);
}
